// PAM_36799279792478
// MI455X (gfx1250) — matrix operand order repaired and checked by reading, not compiled, not run on hardware
//
#include <hip/hip_runtime.h>

// ---------------- problem constants ----------------
constexpr int kB = 4, kC = 64, kH = 64, kW = 256;
constexpr int kD = 48;                      // MAX_DISP band
constexpr int kE = kB * kH * kW * kC;       // elements per feature map (4,194,304)
constexpr float kNEG = -1e8f;

// ---------------- WMMA operand types (CDNA5 wave32) ----------------
typedef __attribute__((ext_vector_type(16))) __bf16 v16bf;
typedef __attribute__((ext_vector_type(8)))  float  v8f;

struct alignas(16) B16 { unsigned u[4]; };   // POD 16-byte chunk (b128 load/store)

union Frag {            // one WMMA A or B operand: 16 bf16 = 32 bytes
  v16bf v;
  B16   b[2];
  unsigned short s[16];
};
union Pack8 {           // 8 bf16 = 16 bytes
  B16 b;
  unsigned short s[8];
};

__device__ __forceinline__ unsigned short f2bf(float f) {
  union { float f; unsigned u; } x; x.f = f;
  unsigned r = x.u + 0x7FFFu + ((x.u >> 16) & 1u);   // round-to-nearest-even
  return (unsigned short)(r >> 16);
}
__device__ __forceinline__ float bf2f(unsigned short h) {
  union { unsigned u; float f; } x; x.u = ((unsigned)h) << 16;
  return x.f;
}

__device__ __forceinline__ float wred_max(float x) {
#pragma unroll
  for (int o = 16; o > 0; o >>= 1) x = fmaxf(x, __shfl_xor(x, o, 32));
  return x;
}
__device__ __forceinline__ float wred_sum(float x) {
#pragma unroll
  for (int o = 16; o > 0; o >>= 1) x += __shfl_xor(x, o, 32);
  return x;
}

// ---- gfx1250 async copy: global -> LDS, 16B per lane, tracked by ASYNCcnt ----
// Pass the REAL shared pointer: ptrtoint escapes the LDS object (so the
// compiler keeps later ds_loads live), and flat shared addresses carry the LDS
// byte offset in addr[31:0] (ISA 10.2), which is what VDST needs.
__device__ __forceinline__ void async_copy16(unsigned short* l, const unsigned short* g) {
  const unsigned loff = (unsigned)(unsigned long long)l;
  asm volatile("global_load_async_to_lds_b128 %0, %1, off"
               :: "v"(loff), "v"(g) : "memory");
}
__device__ __forceinline__ void async_wait0() {
  asm volatile("s_wait_asynccnt 0x0" ::: "memory");
}

// ---------------- layout converters ----------------
// NCHW f32 -> NHWC bf16
__global__ void k_act2bf(const float* __restrict__ in, unsigned short* __restrict__ out) {
  const int idx = blockIdx.x * blockDim.x + threadIdx.x;  // NHWC flat index
  const int c = idx & 63;
  int t = idx >> 6;
  const int x = t & 255; t >>= 8;
  const int y = t & 63;  t >>= 6;
  const int b = t;
  out[idx] = f2bf(in[((size_t)((b * kC + c) * kH + y)) * kW + x]);
}

// OIHW f32 -> [O][(ky*KS+kx)*C + c] bf16  (K-major rows for A-fragments)
__global__ void k_w2bf(const float* __restrict__ in, unsigned short* __restrict__ out, int khw) {
  const int idx = blockIdx.x * blockDim.x + threadIdx.x;
  const int N = kC * kC * khw;
  if (idx >= N) return;
  const int c  = idx % kC;
  const int kk = (idx / kC) % khw;
  const int o  = idx / (kC * khw);
  out[idx] = f2bf(in[(size_t)(o * kC + c) * khw + kk]);
}

// ---------------- implicit-GEMM conv via WMMA bf16 ----------------
// One wave: 16 outchannels x 64 x-positions (4 independent accumulators so the
// bf16 WMMA->WMMA hazard slots are filled with independent WMMAs, no v_nop).
template <int KS, bool RELU, bool RES>
__global__ __launch_bounds__(256) void k_conv(const unsigned short* __restrict__ in,
                                              const unsigned short* __restrict__ wt,
                                              const float* __restrict__ bias,
                                              const unsigned short* __restrict__ res,
                                              unsigned short* __restrict__ out) {
  const int wid  = (blockIdx.x * blockDim.x + threadIdx.x) >> 5;
  const int lane = threadIdx.x & 31;
  const int n  = lane & 15;     // A-row (outchannel) / B-column lane index
  const int hf = lane >> 4;     // lane half per ISA fragment layout

  int t = wid;
  const int ot = t & 3;  t >>= 2;     // outchannel tile (C/16 = 4)
  const int xg = t & 3;  t >>= 2;     // x group of 64 (W/64 = 4)
  const int hh = t & 63; t >>= 6;     // h
  const int bb = t;                   // b

  const int obase = ot << 4;
  const int xbase = xg << 6;
  constexpr int KC = kC * KS * KS;

  const unsigned short* arow0 = wt + (size_t)(obase + n) * KC;

  v8f acc[4] = {};
#pragma unroll
  for (int ky = 0; ky < KS; ++ky) {
    const int y = hh + ky - (KS >> 1);
    const bool yok = (y >= 0) && (y < kH);
    const unsigned short* brow = in + ((size_t)((bb * kH + y) * kW)) * kC;
    if (yok) __builtin_prefetch((const void*)(brow + (size_t)(xbase + 64) * kC), 0, 1);
#pragma unroll
    for (int kx = 0; kx < KS; ++kx) {
      const int dx = kx - (KS >> 1);
      const unsigned short* arow = arow0 + (ky * KS + kx) * kC;
#pragma unroll
      for (int cc = 0; cc < kC; cc += 32) {
        Frag fa;   // A (16x32 bf16): K runs [hf*8..+7] and [16+hf*8..+7]
        fa.b[0] = *(const B16*)(arow + cc + hf * 8);
        fa.b[1] = *(const B16*)(arow + cc + 16 + hf * 8);
#pragma unroll
        for (int u = 0; u < 4; ++u) {   // 4 independent WMMAs per A fragment
          const int xx = xbase + u * 16 + n + dx;
          const bool ok = yok && (xx >= 0) && (xx < kW);
          Frag fb;   // B (32x16 bf16): K runs [hf*8..+7] and [16+hf*8..+7]
          if (ok) {
            const unsigned short* bp = brow + (size_t)xx * kC + cc + hf * 8;
            fb.b[0] = *(const B16*)(bp);
            fb.b[1] = *(const B16*)(bp + 16);
          } else {
            fb.b[0] = B16{0u, 0u, 0u, 0u};
            fb.b[1] = B16{0u, 0u, 0u, 0u};
          }
          acc[u] = __builtin_amdgcn_wmma_f32_16x16x32_bf16(
              false, fa.v, false, fb.v, (short)0, acc[u], false, false);
        }
      }
    }
  }

  float bv[8];
#pragma unroll
  for (int i = 0; i < 8; ++i) bv[i] = bias[obase + hf * 8 + i];

#pragma unroll
  for (int u = 0; u < 4; ++u) {
    // D layout: per lane M = hf*8 + i -> 8 consecutive outchannels, one b128 store
    const size_t oaddr =
        ((size_t)((bb * kH + hh) * kW + xbase + u * 16 + n)) * kC + obase + hf * 8;
    Pack8 rp;
    if (RES) rp.b = *(const B16*)(res + oaddr);
    Pack8 o;
#pragma unroll
    for (int i = 0; i < 8; ++i) {
      float v = acc[u][i] + bv[i];
      if (RES)  v += bf2f(rp.s[i]);
      if (RELU) v = fmaxf(v, 0.f);
      o.s[i] = f2bf(v);
    }
    *(B16*)(out + oaddr) = o.b;
  }
}

// ---------------- attention scores via async-LDS staging + WMMA ----------------
// One workgroup per (b,h): stage Q-row (256x64 bf16) and S-row into LDS with
// GLOBAL_LOAD_ASYNC_TO_LDS_B128, then 8 waves compute the 256x256 score tile
// from LDS. Each wave job = 16 w x 64 v (4 accumulators). Scaled by 1/C.
__global__ __launch_bounds__(256) void k_score(const unsigned short* __restrict__ Q,
                                               const unsigned short* __restrict__ S,
                                               float* __restrict__ M) {
  __shared__ unsigned short lds[2 * kW * kC];   // 64 KB of the 320 KB WGP LDS
  const int bh  = blockIdx.x;                   // b*H + h
  const int tid = threadIdx.x;
  const size_t rowoff = (size_t)bh * kW * kC;
  const unsigned short* Qr = Q + rowoff;
  const unsigned short* Sr = S + rowoff;
#pragma unroll
  for (int i = 0; i < 8; ++i) {                 // 2048 16B chunks per matrix
    const int c = tid + 256 * i;
    async_copy16(&lds[c * 8], Qr + c * 8);
    async_copy16(&lds[kW * kC + c * 8], Sr + c * 8);
  }
  async_wait0();
  __syncthreads();

  const int wave = tid >> 5;
  const int lane = tid & 31;
  const int n = lane & 15, hf = lane >> 4;
  const unsigned short* lq = lds;
  const unsigned short* ls = lds + kW * kC;

  for (int it = 0; it < 8; ++it) {
    const int job = wave + it * 8;              // 64 jobs: 16 w-tiles x 4 v-groups
    const int wt = job & 15;
    const int vg = job >> 4;

    v8f acc[4] = {};
#pragma unroll
    for (int cc = 0; cc < kC; cc += 32) {
      Frag fa;
      const unsigned short* qp = lq + (size_t)(wt * 16 + n) * kC + cc;
      fa.b[0] = *(const B16*)(qp + hf * 8);
      fa.b[1] = *(const B16*)(qp + 16 + hf * 8);
#pragma unroll
      for (int u = 0; u < 4; ++u) {
        Frag fb;
        const unsigned short* sp =
            ls + (size_t)(vg * 64 + u * 16 + n) * kC + cc + hf * 8;
        fb.b[0] = *(const B16*)(sp);
        fb.b[1] = *(const B16*)(sp + 16);
        acc[u] = __builtin_amdgcn_wmma_f32_16x16x32_bf16(
            false, fa.v, false, fb.v, (short)0, acc[u], false, false);
      }
    }
    float* orow = M + ((size_t)bh * kW + wt * 16) * kW + vg * 64 + n;
#pragma unroll
    for (int u = 0; u < 4; ++u)
#pragma unroll
      for (int i = 0; i < 8; ++i) {
        const int m = hf * 8 + i;
        orow[(size_t)m * kW + u * 16] = acc[u][i] * (1.f / 64.f);
      }
  }
}

// ---------------- masked softmax, in place, one wave per (b,h,w) row ----------------
template <bool R2L>
__global__ __launch_bounds__(256) void k_softmax(float* __restrict__ M) {
  const int row  = (blockIdx.x * blockDim.x + threadIdx.x) >> 5;
  const int lane = threadIdx.x & 31;
  const int w = row & (kW - 1);
  float* p = M + (size_t)row * kW;
  float s[8];
#pragma unroll
  for (int j = 0; j < 8; ++j) {
    const int v = lane + 32 * j;
    const bool inval = R2L ? ((v > w) || (v <= w - kD))
                           : ((v < w) || (v >= w + kD));
    s[j] = p[v] + (inval ? kNEG : 0.f);
  }
  float m = s[0];
#pragma unroll
  for (int j = 1; j < 8; ++j) m = fmaxf(m, s[j]);
  m = wred_max(m);
  float e[8], sum = 0.f;
#pragma unroll
  for (int j = 0; j < 8; ++j) { e[j] = __expf(s[j] - m); sum += e[j]; }
  sum = wred_sum(sum);
  const float r = 1.f / (sum + 1e-8f);
#pragma unroll
  for (int j = 0; j < 8; ++j) p[lane + 32 * j] = e[j] * r;
}

// ---------------- V = (column-sum of M_l2r over w) > 0.1 ----------------
__global__ __launch_bounds__(256) void k_valid(const float* __restrict__ Ml2r,
                                               float* __restrict__ V) {
  const int bh = blockIdx.x;
  const int v  = threadIdx.x;
  const float* base = Ml2r + (size_t)bh * kW * kW + v;
  float s = 0.f;
  for (int w = 0; w < kW; ++w) s += base[(size_t)w * kW];
  V[(size_t)bh * kW + v] = (s > 0.1f) ? 1.f : 0.f;
}

// ---------------- disp_init = w - sum_v M_r2l[..,v]*v ----------------
__global__ __launch_bounds__(256) void k_leftpos(const float* __restrict__ Mr2l,
                                                 float* __restrict__ dinit) {
  const int row  = (blockIdx.x * blockDim.x + threadIdx.x) >> 5;
  const int lane = threadIdx.x & 31;
  const int w = row & (kW - 1);
  const float* p = Mr2l + (size_t)row * kW;
  float s = 0.f;
#pragma unroll
  for (int j = 0; j < 8; ++j) { const int v = lane + 32 * j; s += p[v] * (float)v; }
  s = wred_sum(s);
  if (lane == 0) dinit[row] = (float)w - s;
}

// ---------------- sequential hole fill (both directions) in LDS ----------------
__global__ __launch_bounds__(256) void k_fill(const float* __restrict__ dinit,
                                              const float* __restrict__ V,
                                              float* __restrict__ disp) {
  __shared__ float ds[kW];
  __shared__ float vs[kW];
  const int bh = blockIdx.x;
  const int x  = threadIdx.x;
  const float d0 = dinit[(size_t)bh * kW + x];
  const float vd = V[(size_t)bh * kW + x];
  ds[x] = d0 * vd;
  vs[x] = vd;
  __syncthreads();
  for (int it = 0; it < kW; ++it) {          // filter [1,1,0] : neighbor x-1
    const float dm = (x > 0) ? ds[x - 1] : 0.f;
    const float vm = (x > 0) ? vs[x - 1] : 0.f;
    const float dc = ds[x], vc = vs[x];
    const float v1 = vc + vm;
    const float vn = (v1 > 0.f) ? 1.f : 0.f;
    const float dn = dc * vc + (dc + dm) / (v1 + 1e-4f) * (vn - vc);
    __syncthreads();
    ds[x] = dn; vs[x] = vn;
    __syncthreads();
  }
  for (int it = 0; it < kW; ++it) {          // filter [0,1,1] : neighbor x+1
    const float dm = (x < kW - 1) ? ds[x + 1] : 0.f;
    const float vm = (x < kW - 1) ? vs[x + 1] : 0.f;
    const float dc = ds[x], vc = vs[x];
    const float v1 = vc + vm;
    const float vn = (v1 > 0.f) ? 1.f : 0.f;
    const float dn = dc * vc + (dc + dm) / (v1 + 1e-4f) * (vn - vc);
    __syncthreads();
    ds[x] = dn; vs[x] = vn;
    __syncthreads();
  }
  disp[(size_t)bh * kW + x] = d0 * vd + ds[x] * (1.f - vd);
}

// ---------------- driver ----------------
extern "C" void kernel_launch(void* const* d_in, const int* in_sizes, int n_in,
                              void* d_out, int out_size, void* d_ws, size_t ws_size,
                              hipStream_t stream) {
  (void)in_sizes; (void)n_in; (void)out_size; (void)ws_size;
  const float* x_left  = (const float*)d_in[0];
  const float* x_right = (const float*)d_in[1];
  const float* rb_w1 = (const float*)d_in[2];
  const float* rb_b1 = (const float*)d_in[3];
  const float* rb_w2 = (const float*)d_in[4];
  const float* rb_b2 = (const float*)d_in[5];
  const float* b1_w1 = (const float*)d_in[6];
  const float* b1_b1 = (const float*)d_in[7];
  const float* b1_w2 = (const float*)d_in[8];
  const float* b1_b2 = (const float*)d_in[9];
  const float* b2_w1 = (const float*)d_in[10];
  const float* b2_b1 = (const float*)d_in[11];
  const float* b2_w2 = (const float*)d_in[12];
  const float* b2_b2 = (const float*)d_in[13];

  // Output regions: (disp, M_r2l, M_l2r, V) concatenated flat
  float* out    = (float*)d_out;
  float* o_disp = out;
  float* o_Mr2l = out + (size_t)kB * kH * kW;
  float* o_Ml2r = o_Mr2l + (size_t)kB * kH * kW * kW;
  float* o_V    = o_Ml2r + (size_t)kB * kH * kW * kW;

  // Workspace carving (256B aligned)
  char* wsp = (char*)d_ws;
  auto carve = [&](size_t bytes) -> void* {
    void* p = (void*)wsp;
    wsp += (bytes + 255) & ~(size_t)255;
    return p;
  };
  unsigned short* xl = (unsigned short*)carve((size_t)kE * 2);
  unsigned short* xr = (unsigned short*)carve((size_t)kE * 2);
  unsigned short* tb = (unsigned short*)carve((size_t)kE * 2);
  unsigned short* bl = (unsigned short*)carve((size_t)kE * 2);
  unsigned short* br = (unsigned short*)carve((size_t)kE * 2);
  unsigned short* hb = (unsigned short*)carve((size_t)kE * 2);
  unsigned short* Ql = (unsigned short*)carve((size_t)kE * 2);
  unsigned short* Qr = (unsigned short*)carve((size_t)kE * 2);
  unsigned short* Sl = (unsigned short*)carve((size_t)kE * 2);
  unsigned short* Sr = (unsigned short*)carve((size_t)kE * 2);
  unsigned short* wrb1 = (unsigned short*)carve((size_t)kC * kC * 9 * 2);
  unsigned short* wrb2 = (unsigned short*)carve((size_t)kC * kC * 9 * 2);
  unsigned short* wb1a = (unsigned short*)carve((size_t)kC * kC * 9 * 2);
  unsigned short* wb2a = (unsigned short*)carve((size_t)kC * kC * 9 * 2);
  unsigned short* wb1b = (unsigned short*)carve((size_t)kC * kC * 2);
  unsigned short* wb2b = (unsigned short*)carve((size_t)kC * kC * 2);
  float* dinit = (float*)carve((size_t)kB * kH * kW * 4);

  // Layout conversion
  k_act2bf<<<kE / 256, 256, 0, stream>>>(x_left, xl);
  k_act2bf<<<kE / 256, 256, 0, stream>>>(x_right, xr);
  k_w2bf<<<144, 256, 0, stream>>>(rb_w1, wrb1, 9);
  k_w2bf<<<144, 256, 0, stream>>>(rb_w2, wrb2, 9);
  k_w2bf<<<144, 256, 0, stream>>>(b1_w1, wb1a, 9);
  k_w2bf<<<144, 256, 0, stream>>>(b2_w1, wb2a, 9);
  k_w2bf<<<16, 256, 0, stream>>>(b1_w2, wb1b, 1);
  k_w2bf<<<16, 256, 0, stream>>>(b2_w2, wb2b, 1);

  const int convBlocks = kB * kH * (kW / 64) * (kC / 16) * 32 / 256;  // 512
  // resblock on both images (shared weights)
  k_conv<3, true, false><<<convBlocks, 256, 0, stream>>>(xl, wrb1, rb_b1, nullptr, tb);
  k_conv<3, true, true ><<<convBlocks, 256, 0, stream>>>(tb, wrb2, rb_b2, xl, bl);
  k_conv<3, true, false><<<convBlocks, 256, 0, stream>>>(xr, wrb1, rb_b1, nullptr, tb);
  k_conv<3, true, true ><<<convBlocks, 256, 0, stream>>>(tb, wrb2, rb_b2, xr, br);
  // branches: Ql = b1(bl), Sl = b2(bl), Qr = b1(br), Sr = b2(br)
  k_conv<3, true,  false><<<convBlocks, 256, 0, stream>>>(bl, wb1a, b1_b1, nullptr, hb);
  k_conv<1, false, false><<<convBlocks, 256, 0, stream>>>(hb, wb1b, b1_b2, nullptr, Ql);
  k_conv<3, true,  false><<<convBlocks, 256, 0, stream>>>(bl, wb2a, b2_b1, nullptr, hb);
  k_conv<1, false, false><<<convBlocks, 256, 0, stream>>>(hb, wb2b, b2_b2, nullptr, Sl);
  k_conv<3, true,  false><<<convBlocks, 256, 0, stream>>>(br, wb1a, b1_b1, nullptr, hb);
  k_conv<1, false, false><<<convBlocks, 256, 0, stream>>>(hb, wb1b, b1_b2, nullptr, Qr);
  k_conv<3, true,  false><<<convBlocks, 256, 0, stream>>>(br, wb2a, b2_b1, nullptr, hb);
  k_conv<1, false, false><<<convBlocks, 256, 0, stream>>>(hb, wb2b, b2_b2, nullptr, Sr);

  // scores: one workgroup per (b,h), async-LDS staged
  k_score<<<kB * kH, 256, 0, stream>>>(Ql, Sr, o_Mr2l);
  k_score<<<kB * kH, 256, 0, stream>>>(Qr, Sl, o_Ml2r);

  const int rowBlocks = kB * kH * kW * 32 / 256;  // 8192
  k_softmax<true ><<<rowBlocks, 256, 0, stream>>>(o_Mr2l);
  k_softmax<false><<<rowBlocks, 256, 0, stream>>>(o_Ml2r);

  k_valid<<<kB * kH, 256, 0, stream>>>(o_Ml2r, o_V);
  k_leftpos<<<rowBlocks, 256, 0, stream>>>(o_Mr2l, dinit);
  k_fill<<<kB * kH, 256, 0, stream>>>(dinit, o_V, o_disp);
}
